// ELG_A_37993280701010
// MI455X (gfx1250) — matrix operand order repaired and checked by reading, not compiled, not run on hardware
//
#include <hip/hip_runtime.h>
#include <hip/hip_bf16.h>
#include <math.h>

namespace {
constexpr int kE = 512;
constexpr int kH = 8;
constexpr int kHD = 64;
constexpr int kF = 32;
constexpr int kL = 3;
constexpr int kV = 32000;
constexpr int kB = 128;
constexpr int kT = 16;
constexpr float kEPS = 1e-5f;
}

typedef __attribute__((ext_vector_type(16))) __bf16 v16bf;
typedef __attribute__((ext_vector_type(8)))  __bf16 v8bf;
typedef __attribute__((ext_vector_type(8)))  float  v8f;

// ---------------- one-time fp32 -> bf16 conversion (streaming) ----------------
__global__ void f32_to_bf16_kernel(const float* __restrict__ src,
                                   __bf16* __restrict__ dst, int n) {
  for (int i = blockIdx.x * blockDim.x + threadIdx.x; i < n; i += gridDim.x * blockDim.x)
    dst[i] = (__bf16)src[i];
}

// ---------------- Generic GEMM: C(MxN) = A(MxK) @ W(NxK)^T + bias ----------------
// bf16 operands, fp32 accumulate. One wave computes a 16x(16*NT) strip: 1 A tile
// reused across NT B tiles, NT wmma per 32-wide K-chunk. K is a compile-time
// constant so all B-tile addresses are immediate offsets off ONE base pointer
// (keeps global addrspace inference -> global_load_b128, partial loadcnt waits).
//
// A-matrix 16x32 bf16 (ISA 7.12.2): lane m holds row m, K {0..7,16..23} (lanes 0-15)
// or {8..15,24..31} (lanes 16-31). B-matrix 32x16: lane n holds col n = W row n,
// K {0..7,16..23} (lanes 0-15) / {8..15,24..31} (lanes 16-31).
template <int NT, int K, typename OUT>
__global__ void gemm_bias_kernel(const __bf16* __restrict__ A,
                                 const __bf16* __restrict__ W,
                                 const float* __restrict__ bias,
                                 OUT* __restrict__ C,
                                 int M, int N, int relu) {
  const int lane = threadIdx.x & 31;
  const int wid  = blockIdx.x * (blockDim.x >> 5) + (threadIdx.x >> 5);
  const int ngroups = N / (16 * NT);
  const int tiles   = (M >> 4) * ngroups;
  if (wid >= tiles) return;                 // wave-uniform: EXEC all-ones for WMMA
  const int m0 = (wid / ngroups) << 4;      // n-major: block's waves share A rows
  const int n0 = (wid % ngroups) * (16 * NT);

  const int nl = lane & 15;
  const int hi = (lane & 16) ? 1 : 0;

  // single streaming base pointers; tile t of W is at immediate offset t*16*K
  const __bf16* pa = A + (size_t)(m0 + nl) * (size_t)K + (hi ? 8 : 0);
  const __bf16* pw = W + (size_t)(n0 + nl) * (size_t)K + (hi ? 8 : 0);

  v8f acc[NT];
#pragma unroll
  for (int t = 0; t < NT; ++t) acc[t] = (v8f){0.f,0.f,0.f,0.f,0.f,0.f,0.f,0.f};

#pragma unroll 2
  for (int k0 = 0; k0 < K; k0 += 32) {
    // keep the weight stream warm ahead of use (speculative; OOB is dropped)
    __builtin_prefetch(pw + 128, 0, 1);

    const v8bf alo = *(const v8bf*)(pa);
    const v8bf ahi = *(const v8bf*)(pa + 16);
    const v16bf a = __builtin_shufflevector(alo, ahi,
        0, 1, 2, 3, 4, 5, 6, 7, 8, 9, 10, 11, 12, 13, 14, 15);
    v16bf b[NT];
#pragma unroll
    for (int t = 0; t < NT; ++t) {
      const v8bf blo = *(const v8bf*)(pw + (size_t)(t * 16 * K));
      const v8bf bhi = *(const v8bf*)(pw + (size_t)(t * 16 * K) + 16);
      b[t] = __builtin_shufflevector(blo, bhi,
          0, 1, 2, 3, 4, 5, 6, 7, 8, 9, 10, 11, 12, 13, 14, 15);
    }

    pa += 32;
    pw += 32;

#pragma unroll
    for (int t = 0; t < NT; ++t)
      acc[t] = __builtin_amdgcn_wmma_f32_16x16x32_bf16(false, a, false, b[t],
                                                       (short)0, acc[t], false, false);
  }

  // C/D layout: VGPR d -> M = m0 + d (lanes 0-15) or m0 + d + 8 (lanes 16-31); N = lane&15
  const int mo = m0 + (hi ? 8 : 0);
#pragma unroll
  for (int t = 0; t < NT; ++t) {
    const int n = n0 + 16 * t + nl;
    const float bv = bias ? bias[n] : 0.0f;
#pragma unroll
    for (int d = 0; d < 8; ++d) {
      float v = acc[t][d] + bv;
      if (relu) v = fmaxf(v, 0.0f);
      C[(size_t)(mo + d) * (size_t)N + n] = (OUT)v;
    }
  }
}

// ---------------- Self-attention: per (batch, head), seq <= 16, HD = 64 ----------------
// qkv rows: (s*B + b), cols [0:E)=q, [E:2E)=k, [2E:3E)=v, head h at h*64.
// Output written as bf16 (feeds the out-projection GEMM A-side).
__global__ void self_attn_kernel(const float* __restrict__ qkv,
                                 __bf16* __restrict__ attn_o, int seq) {
  const int b = blockIdx.x;
  const int h = blockIdx.y;
  const int tid = threadIdx.x;   // 64 threads (2 waves)
  __shared__ float qs[16 * 64], ks[16 * 64], vs[16 * 64];
  __shared__ float sc[16];
  __shared__ float inv_s;

  for (int idx = tid; idx < seq * 64; idx += 64) {
    const int s = idx >> 6, d = idx & 63;
    const size_t row = (size_t)(s * kB + b) * (size_t)(3 * kE);
    const int col = h * kHD + d;
    qs[idx] = qkv[row + col];
    ks[idx] = qkv[row + kE + col];
    vs[idx] = qkv[row + 2 * kE + col];
  }
  __syncthreads();

  for (int s = 0; s < seq; ++s) {
    if (tid < seq) {
      float acc = 0.f;
#pragma unroll 8
      for (int d = 0; d < 64; ++d) acc += qs[s * 64 + d] * ks[tid * 64 + d];
      sc[tid] = acc * 0.125f;      // 1/sqrt(64)
    }
    __syncthreads();
    if (tid == 0) {
      float mx = sc[0];
      for (int t = 1; t < seq; ++t) mx = fmaxf(mx, sc[t]);
      float sum = 0.f;
      for (int t = 0; t < seq; ++t) { sc[t] = __expf(sc[t] - mx); sum += sc[t]; }
      inv_s = 1.0f / sum;
    }
    __syncthreads();
    float o = 0.f;
    for (int t = 0; t < seq; ++t) o += sc[t] * vs[t * 64 + tid];
    attn_o[(size_t)(s * kB + b) * (size_t)kE + h * kHD + tid] = (__bf16)(o * inv_s);
    __syncthreads();
  }
}

// ---------------- Residual + LayerNorm over E=512, one block per row ----------------
// Writes fp32 x (residual stream) AND bf16 x_bf (GEMM A-side operand).
// per_b != 0: y is indexed by (row % B) (hoisted cross-attention add).
__global__ void residual_ln_kernel(float* __restrict__ x,
                                   __bf16* __restrict__ xbf,
                                   const float* __restrict__ y,
                                   const float* __restrict__ g,
                                   const float* __restrict__ bt,
                                   int per_b) {
  const int row = blockIdx.x;
  const int tid = threadIdx.x;   // 256 threads, 2 elements each
  __shared__ float red[256];
  const float* yr = y + (size_t)(per_b ? (row % kB) : row) * (size_t)kE;
  float* xr = x + (size_t)row * (size_t)kE;
  __bf16* xbr = xbf + (size_t)row * (size_t)kE;

  float v0 = xr[tid] + yr[tid];
  float v1 = xr[tid + 256] + yr[tid + 256];
  red[tid] = v0 + v1;
  __syncthreads();
  for (int s = 128; s > 0; s >>= 1) { if (tid < s) red[tid] += red[tid + s]; __syncthreads(); }
  const float mean = red[0] * (1.0f / kE);
  __syncthreads();

  const float d0 = v0 - mean, d1 = v1 - mean;
  red[tid] = d0 * d0 + d1 * d1;
  __syncthreads();
  for (int s = 128; s > 0; s >>= 1) { if (tid < s) red[tid] += red[tid + s]; __syncthreads(); }
  const float rstd = rsqrtf(red[0] * (1.0f / kE) + kEPS);

  const float r0 = d0 * rstd * g[tid] + bt[tid];
  const float r1 = d1 * rstd * g[tid + 256] + bt[tid + 256];
  xr[tid]        = r0;
  xr[tid + 256]  = r1;
  xbr[tid]       = (__bf16)r0;
  xbr[tid + 256] = (__bf16)r1;
}

// ---------------- Softmax over V=32000 + argmax, one block per batch row ----------------
__global__ void softmax_argmax_kernel(const float* __restrict__ logits,
                                      float* __restrict__ out_probs,
                                      float* __restrict__ out_tok,
                                      int* __restrict__ tok_buf,
                                      int step) {
  const int b = blockIdx.x;
  const int tid = threadIdx.x;   // 256
  __shared__ float rv[256];
  __shared__ int   ri[256];
  const float* lr = logits + (size_t)b * (size_t)kV;

  float mx = -3.4e38f; int mi = 0;
  for (int j = tid; j < kV; j += 256) {
    const float v = lr[j];
    if (v > mx) { mx = v; mi = j; }
  }
  rv[tid] = mx; ri[tid] = mi;
  __syncthreads();
  for (int s = 128; s > 0; s >>= 1) {
    if (tid < s) {
      const float a = rv[tid], c = rv[tid + s];
      if (c > a || (c == a && ri[tid + s] < ri[tid])) { rv[tid] = c; ri[tid] = ri[tid + s]; }
    }
    __syncthreads();
  }
  const float gmax = rv[0];
  const int   gidx = ri[0];
  __syncthreads();

  float sum = 0.f;
  for (int j = tid; j < kV; j += 256) sum += __expf(lr[j] - gmax);
  rv[tid] = sum;
  __syncthreads();
  for (int s = 128; s > 0; s >>= 1) { if (tid < s) rv[tid] += rv[tid + s]; __syncthreads(); }
  const float inv = 1.0f / rv[0];

  float* pr = out_probs + (size_t)b * (size_t)kV;
  for (int j = tid; j < kV; j += 256) pr[j] = __expf(lr[j] - gmax) * inv;
  if (tid == 0) {
    tok_buf[b] = gidx;
    out_tok[(size_t)b * kT + step] = (float)gidx;   // spoken_token is (B, T)
  }
}

// ---------------- Positional encoding table (T x E) ----------------
__global__ void pos_enc_kernel(float* __restrict__ pe) {
  const int t = blockIdx.x;
  for (int e = threadIdx.x; e < kE; e += 256) {
    const int j = e >> 1;
    const float div = __expf((float)(2 * j) * (-logf(10000.0f) / (float)kE));
    const float a = (float)t * div;
    pe[(size_t)t * kE + e] = (e & 1) ? __cosf(a) : __sinf(a);
  }
}

// ---------------- x[0, b, :] = emb[SOS] (fp32 + bf16) ----------------
__global__ void init_x0_kernel(float* __restrict__ x, __bf16* __restrict__ xbf,
                               const float* __restrict__ emb) {
  const int b = blockIdx.x;
  for (int e = threadIdx.x; e < kE; e += 256) {
    const float v = emb[(size_t)1 * kE + e];   // SOS = 1
    x[(size_t)b * kE + e]   = v;
    xbf[(size_t)b * kE + e] = (__bf16)v;
  }
}

// ---------------- Append next token embedding + pe[step] (fp32 + bf16) ----------------
__global__ void append_kernel(float* __restrict__ x, __bf16* __restrict__ xbf,
                              const float* __restrict__ emb,
                              const float* __restrict__ pe, const int* __restrict__ tok,
                              int step) {
  const int b = blockIdx.x;
  const int t = tok[b];
  const size_t base = (size_t)(step + 1) * kB * kE + (size_t)b * kE;
  for (int e = threadIdx.x; e < kE; e += 256) {
    const float v = emb[(size_t)t * kE + e] + pe[(size_t)step * kE + e];
    x[base + e]   = v;
    xbf[base + e] = (__bf16)v;
  }
}

extern "C" void kernel_launch(void* const* d_in, const int* in_sizes, int n_in,
                              void* d_out, int out_size, void* d_ws, size_t ws_size,
                              hipStream_t stream) {
  const float* cur_room = (const float*)d_in[0];
  const float* emb      = (const float*)d_in[1];
  const float* sa_in_w  = (const float*)d_in[2];
  const float* sa_in_b  = (const float*)d_in[3];
  const float* sa_out_w = (const float*)d_in[4];
  const float* sa_out_b = (const float*)d_in[5];
  const float* ca_in_w  = (const float*)d_in[6];
  const float* ca_in_b  = (const float*)d_in[7];
  const float* ca_out_w = (const float*)d_in[8];
  const float* ca_out_b = (const float*)d_in[9];
  const float* ff1_w    = (const float*)d_in[10];
  const float* ff1_b    = (const float*)d_in[11];
  const float* ff2_w    = (const float*)d_in[12];
  const float* ff2_b    = (const float*)d_in[13];
  const float* ln_g     = (const float*)d_in[14];
  const float* ln_b     = (const float*)d_in[15];
  const float* out_w    = (const float*)d_in[16];
  const float* out_b    = (const float*)d_in[17];
  (void)in_sizes; (void)n_in; (void)out_size; (void)ws_size;

  // ---- workspace layout: fp32 region, then bf16 region, then tok ----
  float* ws     = (float*)d_ws;
  float* x      = ws;                                    // (T+1)*B*E
  float* qkv    = x      + (size_t)(kT + 1) * kB * kE;   // T*B*3E
  float* tmp    = qkv    + (size_t)kT * kB * 3 * kE;     // T*B*E
  float* logits = tmp    + (size_t)kT * kB * kE;         // B*V
  float* ca_add = logits + (size_t)kB * kV;              // L*B*E
  float* pe     = ca_add + (size_t)kL * kB * kE;         // T*E

  __bf16* bfbase    = (__bf16*)(pe + (size_t)kT * kE);
  __bf16* x_bf      = bfbase;                                    // (T+1)*B*E
  __bf16* attn_o_bf = x_bf      + (size_t)(kT + 1) * kB * kE;    // T*B*E
  __bf16* ffh_bf    = attn_o_bf + (size_t)kT * kB * kE;          // T*B*F
  __bf16* room_bf   = ffh_bf    + (size_t)kT * kB * kF;          // B*E
  __bf16* ca_tmp_bf = room_bf   + (size_t)kB * kE;               // B*E
  __bf16* ca_wv_bf  = ca_tmp_bf + (size_t)kB * kE;               // L*E*E
  __bf16* ca_ow_bf  = ca_wv_bf  + (size_t)kL * kE * kE;          // L*E*E
  __bf16* sa_in_bf  = ca_ow_bf  + (size_t)kL * kE * kE;          // L*3E*E
  __bf16* sa_ow_bf  = sa_in_bf  + (size_t)kL * 3 * kE * kE;      // L*E*E
  __bf16* ff1_bf    = sa_ow_bf  + (size_t)kL * kE * kE;          // L*F*E
  __bf16* ff2_bf    = ff1_bf    + (size_t)kL * kF * kE;          // L*E*F
  __bf16* out_w_bf  = ff2_bf    + (size_t)kL * kE * kF;          // V*E
  int*    tok       = (int*)(out_w_bf + (size_t)kV * kE);        // B ints

  float* out = (float*)d_out;   // [B*T tokens][T*B*V probs]

  auto cvt = [&](const float* s, __bf16* d, size_t n) {
    const int blocks = (int)((n + 1023) / 1024);
    f32_to_bf16_kernel<<<dim3(blocks), dim3(256), 0, stream>>>(s, d, (int)n);
  };
  // K = 512 GEMMs
  auto gemm512 = [&](const __bf16* A, const __bf16* W, const float* bias, float* C,
                     int M, int N, int relu) {
    if (N % 64 == 0) {
      const int tiles = (M / 16) * (N / 64);
      gemm_bias_kernel<4, 512, float><<<dim3((tiles + 7) / 8), dim3(256), 0, stream>>>(
          A, W, bias, C, M, N, relu);
    } else {
      const int tiles = (M / 16) * (N / 32);
      gemm_bias_kernel<2, 512, float><<<dim3((tiles + 7) / 8), dim3(256), 0, stream>>>(
          A, W, bias, C, M, N, relu);
    }
  };
  auto gemm512_bf = [&](const __bf16* A, const __bf16* W, const float* bias, __bf16* C,
                        int M, int N, int relu) {
    if (N % 64 == 0) {
      const int tiles = (M / 16) * (N / 64);
      gemm_bias_kernel<4, 512, __bf16><<<dim3((tiles + 7) / 8), dim3(256), 0, stream>>>(
          A, W, bias, C, M, N, relu);
    } else {
      const int tiles = (M / 16) * (N / 32);
      gemm_bias_kernel<2, 512, __bf16><<<dim3((tiles + 7) / 8), dim3(256), 0, stream>>>(
          A, W, bias, C, M, N, relu);
    }
  };
  // K = 32 GEMM (ff2 expand), N = 512
  auto gemm32 = [&](const __bf16* A, const __bf16* W, const float* bias, float* C,
                    int M, int N, int relu) {
    const int tiles = (M / 16) * (N / 64);
    gemm_bias_kernel<4, 32, float><<<dim3((tiles + 7) / 8), dim3(256), 0, stream>>>(
        A, W, bias, C, M, N, relu);
  };

  // ---- one-time bf16 conversion of every GEMM operand ----
  cvt(sa_in_w, sa_in_bf, (size_t)kL * 3 * kE * kE);
  cvt(sa_out_w, sa_ow_bf, (size_t)kL * kE * kE);
  cvt(ca_out_w, ca_ow_bf, (size_t)kL * kE * kE);
  cvt(ff1_w, ff1_bf, (size_t)kL * kF * kE);
  cvt(ff2_w, ff2_bf, (size_t)kL * kE * kF);
  cvt(out_w, out_w_bf, (size_t)kV * kE);
  cvt(cur_room, room_bf, (size_t)kB * kE);
  for (int l = 0; l < kL; ++l)   // wv slice of ca_in_w: rows [2E:3E)
    cvt(ca_in_w + (size_t)l * 3 * kE * kE + (size_t)2 * kE * kE,
        ca_wv_bf + (size_t)l * kE * kE, (size_t)kE * kE);

  pos_enc_kernel<<<kT, 256, 0, stream>>>(pe);
  init_x0_kernel<<<kB, 256, 0, stream>>>(x, x_bf, emb);

  // Cross-attention collapses (memory seq-len == 1 -> softmax == 1):
  //   ca_add[l][b] = (hx[b] @ wv_l^T + bv_l) @ ca_out_w_l^T + ca_out_b_l   (step-invariant)
  for (int l = 0; l < kL; ++l) {
    gemm512_bf(room_bf, ca_wv_bf + (size_t)l * kE * kE,
               ca_in_b + (size_t)l * 3 * kE + 2 * kE, ca_tmp_bf, kB, kE, 0);
    gemm512(ca_tmp_bf, ca_ow_bf + (size_t)l * kE * kE, ca_out_b + (size_t)l * kE,
            ca_add + (size_t)l * kB * kE, kB, kE, 0);
  }

  for (int i = 0; i < kT; ++i) {
    const int seq  = i + 1;
    const int rows = seq * kB;
    for (int l = 0; l < kL; ++l) {
      const float* g0 = ln_g + ((size_t)l * 3 + 0) * kE; const float* b0 = ln_b + ((size_t)l * 3 + 0) * kE;
      const float* g1 = ln_g + ((size_t)l * 3 + 1) * kE; const float* b1 = ln_b + ((size_t)l * 3 + 1) * kE;
      const float* g2 = ln_g + ((size_t)l * 3 + 2) * kE; const float* b2 = ln_b + ((size_t)l * 3 + 2) * kE;

      // self-attention
      gemm512(x_bf, sa_in_bf + (size_t)l * 3 * kE * kE, sa_in_b + (size_t)l * 3 * kE,
              qkv, rows, 3 * kE, 0);
      self_attn_kernel<<<dim3(kB, kH), 64, 0, stream>>>(qkv, attn_o_bf, seq);
      gemm512(attn_o_bf, sa_ow_bf + (size_t)l * kE * kE, sa_out_b + (size_t)l * kE,
              tmp, rows, kE, 0);
      residual_ln_kernel<<<rows, 256, 0, stream>>>(x, x_bf, tmp, g0, b0, 0);

      // cross-attention: hoisted per-batch additive
      residual_ln_kernel<<<rows, 256, 0, stream>>>(x, x_bf, ca_add + (size_t)l * kB * kE, g1, b1, 1);

      // FFN (F=32): relu GEMM (bf16 out, N=32 -> NT=2) then expand GEMM (K=32)
      gemm512_bf(x_bf, ff1_bf + (size_t)l * kF * kE, ff1_b + (size_t)l * kF,
                 ffh_bf, rows, kF, 1);
      gemm32(ffh_bf, ff2_bf + (size_t)l * kE * kF, ff2_b + (size_t)l * kE,
             tmp, rows, kE, 0);
      residual_ln_kernel<<<rows, 256, 0, stream>>>(x, x_bf, tmp, g2, b2, 0);
    }

    // vocab projection on last position only: (128 x 32000 x 512) — the hot GEMM
    gemm512(x_bf + (size_t)(seq - 1) * kB * kE, out_w_bf, out_b, logits, kB, kV, 0);
    softmax_argmax_kernel<<<kB, 256, 0, stream>>>(
        logits, out + (size_t)kB * kT + (size_t)i * kB * kV, out, tok, i);
    if (i + 1 < kT)
      append_kernel<<<kB, 256, 0, stream>>>(x, x_bf, emb, pe, tok, i);
  }
}
